// Model_38732015075463
// MI455X (gfx1250) — matrix operand order repaired and checked by reading, not compiled, not run on hardware
//
#include <hip/hip_runtime.h>
#include <hip/hip_bf16.h>

// ---------------------------------------------------------------------------
// Flash-attention forward, B=2,H=8,S=2048,D=15 (fp32 in/out), gfx1250.
// Compute-bound => bf16 WMMA (v_wmma_f32_16x16x32_bf16) for QK^T and P*V.
// Block = 8 waves; wave = one 16-query tile; keys processed 64 at a time.
// K/V tiles double-buffered in LDS via GLOBAL_LOAD_ASYNC_TO_LDS_B128
// (ASYNCcnt) so DMA overlaps compute; softmax in exp2 domain.
// ---------------------------------------------------------------------------

typedef __attribute__((ext_vector_type(16))) __bf16 v16bf;
typedef __attribute__((ext_vector_type(8)))  float  v8f;

#define S_LEN 2048
#define BH    16          // B*H
#define DHEAD 15
#define DPAD  16
#define KEEP_BYTE 77u     // ~0.30 * 256 (keep prob)
#define INV_KEEP 3.3333333333f

__device__ __forceinline__ unsigned hash_u32(unsigned x) {
    x *= 2654435761u; x ^= x >> 16;
    x *= 2246822519u; x ^= x >> 13;
    return x;
}

// --- pre-pass: f32 -> bf16, pad D 15->16 for Q/K, transpose V to [bh][d][s] ---
__global__ void attn_prepass(const float* __restrict__ Q,
                             const float* __restrict__ K,
                             const float* __restrict__ V,
                             __bf16* __restrict__ Qp,
                             __bf16* __restrict__ Kp,
                             __bf16* __restrict__ Vt) {
    int idx = blockIdx.x * blockDim.x + threadIdx.x;     // over BH*S*16
    if (idx >= BH * S_LEN * DPAD) return;
    int d  = idx & (DPAD - 1);
    int bs = idx >> 4;                                   // bh*S + s
    bool valid = (d < DHEAD);
    float q = valid ? Q[bs * DHEAD + d] : 0.0f;
    float k = valid ? K[bs * DHEAD + d] : 0.0f;
    float v = valid ? V[bs * DHEAD + d] : 0.0f;
    Qp[idx] = (__bf16)q;
    Kp[idx] = (__bf16)k;
    int s  = bs & (S_LEN - 1);
    int bh = bs >> 11;                                   // S_LEN = 2^11
    Vt[((size_t)(bh * DPAD + d)) * S_LEN + s] = (__bf16)v;
}

// --- main attention kernel: 8 waves/block, 1 wave = 16 query rows,
//     64-key steps, double-buffered LDS staging ---
__global__ __launch_bounds__(256) void attn_fwd(
    const __bf16* __restrict__ Qp, const __bf16* __restrict__ Kp,
    const __bf16* __restrict__ Vt, const float* __restrict__ mask,
    float* __restrict__ out)
{
    // LDS (bf16 elems):
    //   buf b in {0,1}:  K tile 64x16 at b*2048,  V tile 16x64 at b*2048+1024
    //   per-wave P tile 16x64 at 4096 + wave*1024
    __shared__ __bf16 lds[2 * 2048 + 8 * 1024];
    const int wave = threadIdx.x >> 5;
    __bf16* ldsP = lds + 4096 + wave * 1024;   // [row][key]

    const int lane = threadIdx.x & 31;
    const int half = lane >> 4;
    const int l16  = lane & 15;
    const int bh   = blockIdx.y;
    const int q0   = blockIdx.x * 128 + wave * 16;
    const float SCALE2 = 0.25819888974716113f * 1.4426950408889634f; // scale*log2e
    const float LOG2E  = 1.4426950408889634f;

    // ---- Q tile in WMMA-A layout: lane row M=l16; half selects d 0..7 / 8..15
    v16bf aq;
    #pragma unroll
    for (int i = 0; i < 16; ++i) aq[i] = (__bf16)0.0f;
    ((uint4*)&aq)[0] = *(const uint4*)(Qp +
        ((size_t)(bh * S_LEN + q0 + l16)) * DPAD + 8 * half);

    v8f o;
    float mrow[8], lrow[8];
    #pragma unroll
    for (int r = 0; r < 8; ++r) { o[r] = 0.0f; mrow[r] = -1e30f; lrow[r] = 0.0f; }

    // ---- per-thread async staging assignment (1 x b128 per thread per step)
    // waves 0-3 stage K (64 rows x 32B), waves 4-7 stage V (16 rows x 128B)
    const int t = threadIdx.x;
    const __bf16* gptr;
    unsigned lbyte;
    int gstep;
    if (t < 128) {
        int row = t >> 1, c = t & 1;
        gptr  = Kp + ((size_t)(bh * S_LEN) + row) * DPAD + c * 8;
        gstep = 64 * DPAD;
        lbyte = (unsigned)(row * 32 + c * 16);
    } else {
        int u = t - 128;
        int d = u >> 3, c = u & 7;
        gptr  = Vt + ((size_t)(bh * DPAD + d)) * S_LEN + c * 8;
        gstep = 64;
        lbyte = (unsigned)(2048 + d * 128 + c * 16);
    }
    // flat LDS pointer low 32 bits == LDS byte offset (aperture is in hi 32)
    lbyte += (unsigned)(size_t)(const void*)lds;

    // ---- prologue: stage tile 0 into buffer 0
    asm volatile("global_load_async_to_lds_b128 %0, %1, off"
                 :: "v"(lbyte), "v"(gptr) : "memory");
    gptr += gstep;

    int buf = 0;
    for (int j = 0; j < S_LEN; j += 64) {
        // ---- current tile landed (own loads), publish to all waves ----
        asm volatile("s_wait_asynccnt 0" ::: "memory");
        __syncthreads();

        // ---- kick off next tile into the other buffer; overlaps compute ----
        if (j + 64 < S_LEN) {
            asm volatile("global_load_async_to_lds_b128 %0, %1, off"
                         :: "v"(lbyte + (unsigned)((buf ^ 1) * 4096)), "v"(gptr)
                         : "memory");
            gptr += gstep;
        }

        const __bf16* ldsK = lds + buf * 2048;          // [key][d]
        const __bf16* ldsV = lds + buf * 2048 + 1024;   // [d][key]

        // ---- 4 x QK^T WMMA (16 keys each); convert to log2-domain scores
        //      immediately to keep accumulator live ranges short ----
        v8f zc;
        #pragma unroll
        for (int r = 0; r < 8; ++r) zc[r] = 0.0f;
        const float* mbase = mask + (size_t)(q0 + 8 * half) * S_LEN + j + l16;
        __builtin_prefetch(mbase + 64, 0, 1);   // next step's mask columns

        float s[4][8];
        #pragma unroll
        for (int tile = 0; tile < 4; ++tile) {
            v16bf bk;
            #pragma unroll
            for (int i = 0; i < 16; ++i) bk[i] = (__bf16)0.0f;
            {   // half selects d 0..7 / 8..15 in elems 0..7; elems 8..15 are zero-pad
                const uint4* kp = (const uint4*)(ldsK + (tile * 16 + l16) * DPAD);
                ((uint4*)&bk)[0] = kp[half];
            }
            v8f c = __builtin_amdgcn_wmma_f32_16x16x32_bf16(
                false, aq, false, bk, (short)0, zc, false, false);
            #pragma unroll
            for (int r = 0; r < 8; ++r)
                s[tile][r] = c[r] * SCALE2 + mbase[(size_t)r * S_LEN + tile * 16] * LOG2E;
        }

        // ---- online softmax (exp2 domain) + dropout; stage P to LDS ----
        #pragma unroll
        for (int r = 0; r < 8; ++r) {
            float rm = fmaxf(fmaxf(s[0][r], s[1][r]), fmaxf(s[2][r], s[3][r]));
            rm = fmaxf(rm, __shfl_xor(rm, 1, 32));
            rm = fmaxf(rm, __shfl_xor(rm, 2, 32));
            rm = fmaxf(rm, __shfl_xor(rm, 4, 32));
            rm = fmaxf(rm, __shfl_xor(rm, 8, 32));
            float mnew = fmaxf(mrow[r], rm);
            float corr = __builtin_amdgcn_exp2f(mrow[r] - mnew);
            float p[4];
            #pragma unroll
            for (int tile = 0; tile < 4; ++tile)
                p[tile] = __builtin_amdgcn_exp2f(s[tile][r] - mnew);
            float rs = (p[0] + p[1]) + (p[2] + p[3]);
            rs += __shfl_xor(rs, 1, 32);
            rs += __shfl_xor(rs, 2, 32);
            rs += __shfl_xor(rs, 4, 32);
            rs += __shfl_xor(rs, 8, 32);
            lrow[r] = lrow[r] * corr + rs;    // softmax denom: no dropout
            mrow[r] = mnew;
            o[r] *= corr;

            // one hash -> 4 keep decisions (one byte per 16-key tile)
            int grow = q0 + 8 * half + r;
            unsigned hx = hash_u32(
                (unsigned)((bh * S_LEN + grow) * S_LEN + j + l16) ^ 0x9E3779B9u);
            int lr = 8 * half + r;
            #pragma unroll
            for (int tile = 0; tile < 4; ++tile) {
                float pd = (((hx >> (8 * tile)) & 0xFFu) < KEEP_BYTE)
                               ? p[tile] * INV_KEEP : 0.0f;
                ldsP[lr * 64 + tile * 16 + l16] = (__bf16)pd;
            }
        }
        asm volatile("s_wait_dscnt 0" ::: "memory");

        // ---- 2 x P*V WMMA (32 keys each): P from LDS in A layout, V B layout
        #pragma unroll
        for (int g = 0; g < 2; ++g) {
            v16bf pa, bv;
            const uint4* pp0 = (const uint4*)(ldsP + l16 * 64 + g * 32 + 8 * half);
            const uint4* pp1 = (const uint4*)(ldsP + l16 * 64 + g * 32 + 16 + 8 * half);
            ((uint4*)&pa)[0] = pp0[0];
            ((uint4*)&pa)[1] = pp1[0];
            const uint4* vp = (const uint4*)(ldsV + l16 * 64 + g * 32 + 8 * half);
            ((uint4*)&bv)[0] = vp[0];
            ((uint4*)&bv)[1] = vp[2];
            o = __builtin_amdgcn_wmma_f32_16x16x32_bf16(
                false, pa, false, bv, (short)0, o, false, false);
        }
        buf ^= 1;
        // no trailing barrier: next step's entry barrier (after its own
        // s_wait_asynccnt) guarantees every wave has finished reading this
        // buffer before it can be re-issued (overwrite targets buf^1 only).
    }

    // ---- normalize and store (output D=15, unpadded)
    #pragma unroll
    for (int r = 0; r < 8; ++r) {
        float val = o[r] / lrow[r];
        int grow = q0 + 8 * half + r;
        if (l16 < DHEAD)
            out[(size_t)(bh * S_LEN + grow) * DHEAD + l16] = val;
    }
}

extern "C" void kernel_launch(void* const* d_in, const int* in_sizes, int n_in,
                              void* d_out, int out_size, void* d_ws, size_t ws_size,
                              hipStream_t stream) {
    const float* Q    = (const float*)d_in[0];
    const float* K    = (const float*)d_in[1];
    const float* V    = (const float*)d_in[2];
    const float* mask = (const float*)d_in[3];
    float* out = (float*)d_out;

    const size_t tile_elems = (size_t)BH * S_LEN * DPAD;   // 524288
    __bf16* Qp = (__bf16*)d_ws;
    __bf16* Kp = Qp + tile_elems;
    __bf16* Vt = Kp + tile_elems;

    int threads = 256;
    int blocks = (int)((tile_elems + threads - 1) / threads);
    attn_prepass<<<blocks, threads, 0, stream>>>(Q, K, V, Qp, Kp, Vt);

    dim3 grid(S_LEN / 128, BH);   // 16 x 16 blocks, 8 waves each
    attn_fwd<<<grid, 256, 0, stream>>>(Qp, Kp, Vt, mask, out);
}
